// LSTM_13786845020299
// MI455X (gfx1250) — matrix operand order repaired and checked by reading, not compiled, not run on hardware
//
#include <hip/hip_runtime.h>
#include <hip/hip_bf16.h>
#include <stdint.h>

// CDNA5 (gfx1250) wave32 WMMA LSTM.
// Phase 1: xW = x @ Wx + b  via v_wmma_f32_16x16x32_bf16, TDM tile staging.
// Phase 2: persistent 32-WG kernel; Wh slice held in VGPR B-fragments for all
//          1024 steps; h broadcast staged by TENSOR_LOAD_TO_LDS each step.

typedef __attribute__((ext_vector_type(16))) __bf16 v16bf;
typedef __attribute__((ext_vector_type(8)))  float   v8f;

union BF32B { v16bf v; uint4 q[2]; };   // 32-byte WMMA A/B fragment

__device__ __forceinline__ float sigmf(float x) { return 1.0f / (1.0f + __expf(-x)); }

// ---------------------------------------------------------------- TDM
#if defined(__has_builtin)
#if __has_builtin(__builtin_amdgcn_tensor_load_to_lds) && \
    __has_builtin(__builtin_amdgcn_s_wait_tensorcnt)
#define USE_TDM 1
#endif
#endif
#ifndef USE_TDM
#define USE_TDM 0
#endif

#if USE_TDM
typedef __attribute__((ext_vector_type(4))) unsigned int u32x4;
typedef __attribute__((ext_vector_type(4))) int          i32x4;
typedef __attribute__((ext_vector_type(8))) int          i32x8;

// D# group1 dword0 flag packs: data_size=1 (2B), pad_enable=1
//   KF1: pad after 64B  (interval code 3), pad 4 DWORDs (code 3)  -> LDS stride 80B
//   KF2: pad after 1024B (interval code 7), pad 4 DWORDs (code 3) -> LDS stride 1040B
#define TDM_KF1 ((1u << 16) | (1u << 20) | (3u << 22) | (3u << 25))
#define TDM_KF2 ((1u << 16) | (1u << 20) | (7u << 22) | (3u << 25))

__device__ __forceinline__ uint32_t lds_off(const void* p) {
  return (uint32_t)(uintptr_t)p;   // generic LDS ptr: offset lives in addr[31:0]
}

// 2D tile load: tile == tensor (never OOB). tile_d0 elems/row, `rows` rows,
// row stride `stride0` elems, bf16 elements.
__device__ __forceinline__ void tdm_load_2d(uint64_t g, uint32_t lds,
                                            uint32_t tile_d0, uint32_t rows,
                                            uint32_t stride0, uint32_t flags) {
  u32x4 g0;
  g0[0] = 1u;                                              // count=1, user D#
  g0[1] = lds;                                             // lds_addr
  g0[2] = (uint32_t)g;                                     // global_addr[31:0]
  g0[3] = (uint32_t)((g >> 32) & 0x1FFFFFFull) | (2u << 30); // addr[56:32]|type=2
  i32x8 g1;
  g1[0] = (int)flags;                                      // mask=0|dsize|pad cfg
  g1[1] = (int)((tile_d0 & 0xFFFFu) << 16);                // tensor_dim0 lo
  g1[2] = (int)(((tile_d0 >> 16) & 0xFFFFu) | ((rows & 0xFFFFu) << 16)); // d0 hi|d1 lo
  g1[3] = (int)(((rows >> 16) & 0xFFFFu) | ((tile_d0 & 0xFFFFu) << 16)); // d1 hi|tile_d0
  g1[4] = (int)(rows & 0xFFFFu);                           // tile_dim1 | tile_dim2=0
  g1[5] = (int)stride0;                                    // tensor_dim0_stride lo
  g1[6] = 0;                                               // stride0 hi | stride1 lo
  g1[7] = 0;
  i32x4 z4 = (i32x4)0;
#if __has_include(<hip/amd_detail/amd_gfx1250_TDM.h>)
  i32x8 z8 = (i32x8)0;
  __builtin_amdgcn_tensor_load_to_lds(g0, g1, z4, z4, z8, 0);
#else
  __builtin_amdgcn_tensor_load_to_lds(g0, g1, z4, z4, 0);
#endif
}
#endif  // USE_TDM

// ---------------------------------------------------------------- converts
__global__ void k_cvt_bf16(const float* __restrict__ s, __bf16* __restrict__ d, int n) {
  int i = blockIdx.x * 256 + threadIdx.x;
  if (i < n) d[i] = (__bf16)s[i];
}

// dst[col*512 + k] = src[k*2048 + col]  (512x2048 f32 -> 2048x512 bf16, K-contiguous)
__global__ void k_transpose512x2048(const float* __restrict__ s, __bf16* __restrict__ d) {
  int i = blockIdx.x * 256 + threadIdx.x;  // exactly 2048*512 threads
  int k = i & 511, col = i >> 9;
  d[i] = (__bf16)s[k * 2048 + col];
}

// ---------------------------------------------------------------- phase 1
// Grid (16, 256): 128x128 output tile per WG, 8 waves.
// Wave grid 4(M) x 2(N); each wave: 2x4 subtiles of 16x16, K-chunks of 32.
// Tiles double-buffered in LDS; staged by TDM (wave 0) overlapped with WMMA.
__global__ __launch_bounds__(256) void k_gemm_xw(
    const __bf16* __restrict__ xb, const __bf16* __restrict__ WxT,
    const float* __restrict__ bias, float* __restrict__ xW) {
  __shared__ __bf16 As[2][128 * 40];  // row-major [row][k], stride 40 (TDM pad)
  __shared__ __bf16 Bs[2][128 * 40];  // transposed [col][k], stride 40
  const int tid  = threadIdx.x;
  const int lane = tid & 31, w = tid >> 5;
  const int ln = lane & 15, hi = lane >> 4;
  const int wm = w >> 1;      // 0..3 : 32-row band
  const int wn = w & 1;       // 0..1 : 64-col band
  const int cb = blockIdx.x;  // 0..15
  const int rb = blockIdx.y;  // 0..255

  v8f acc[2][4];
#pragma unroll
  for (int sm = 0; sm < 2; ++sm)
#pragma unroll
    for (int sn = 0; sn < 4; ++sn) {
      float bv = bias[cb * 128 + wn * 64 + sn * 16 + ln];
#pragma unroll
      for (int rr = 0; rr < 8; ++rr) acc[sm][sn][rr] = bv;
    }

#if USE_TDM
  const uint64_t gA = (uint64_t)(uintptr_t)(xb + (size_t)(rb * 128) * 512);
  const uint64_t gB = (uint64_t)(uintptr_t)(WxT + (size_t)(cb * 128) * 512);
  const uint32_t la0 = lds_off(&As[0][0]), la1 = lds_off(&As[1][0]);
  const uint32_t lb0 = lds_off(&Bs[0][0]), lb1 = lds_off(&Bs[1][0]);
  if (w == 0) {
    tdm_load_2d(gA, la0, 32, 128, 512, TDM_KF1);
    tdm_load_2d(gB, lb0, 32, 128, 512, TDM_KF1);
    __builtin_amdgcn_s_wait_tensorcnt(0);
  }
  __syncthreads();
#endif
  const int srow = tid >> 1, shalf = tid & 1;
  for (int kc = 0; kc < 16; ++kc) {
    const int cur = kc & 1;
#if USE_TDM
    if (w == 0 && kc < 15) {  // prefetch next chunk into other buffer
      uint64_t off = (uint64_t)(kc + 1) * 64u;
      tdm_load_2d(gA + off, cur ? la0 : la1, 32, 128, 512, TDM_KF1);
      tdm_load_2d(gB + off, cur ? lb0 : lb1, 32, 128, 512, TDM_KF1);
    }
#else
    {  // cooperative stage: 128 rows x 32 elems, 2 threads/row x 32B
      const uint4* ga = (const uint4*)(xb + (size_t)(rb * 128 + srow) * 512 + kc * 32 + shalf * 16);
      uint4* la = (uint4*)(As[cur] + srow * 40 + shalf * 16);
      la[0] = ga[0]; la[1] = ga[1];
      const uint4* gb = (const uint4*)(WxT + (size_t)(cb * 128 + srow) * 512 + kc * 32 + shalf * 16);
      uint4* lb = (uint4*)(Bs[cur] + srow * 40 + shalf * 16);
      lb[0] = gb[0]; lb[1] = gb[1];
    }
    __syncthreads();
#endif
    BF32B a[2], b[4];
#pragma unroll
    for (int sm = 0; sm < 2; ++sm) {
      const __bf16* p = As[cur] + (wm * 32 + sm * 16 + ln) * 40 + hi * 8;
      a[sm].q[0] = *(const uint4*)p;
      a[sm].q[1] = *(const uint4*)(p + 16);
    }
#pragma unroll
    for (int sn = 0; sn < 4; ++sn) {
      const __bf16* p = Bs[cur] + (wn * 64 + sn * 16 + ln) * 40 + hi * 8;
      b[sn].q[0] = *(const uint4*)p;
      b[sn].q[1] = *(const uint4*)(p + 16);
    }
#pragma unroll
    for (int sm = 0; sm < 2; ++sm)
#pragma unroll
      for (int sn = 0; sn < 4; ++sn)
        acc[sm][sn] = __builtin_amdgcn_wmma_f32_16x16x32_bf16(
            false, a[sm].v, false, b[sn].v, (short)0, acc[sm][sn], false, false);
#if USE_TDM
    if (w == 0) __builtin_amdgcn_s_wait_tensorcnt(0);
#endif
    __syncthreads();
  }
#pragma unroll
  for (int sm = 0; sm < 2; ++sm)
#pragma unroll
    for (int sn = 0; sn < 4; ++sn) {
      int colg = cb * 128 + wn * 64 + sn * 16 + ln;
      size_t rg = (size_t)(rb * 128 + wm * 32 + sm * 16 + hi * 8);
#pragma unroll
      for (int rr = 0; rr < 8; ++rr)
        xW[(rg + rr) * 2048 + colg] = acc[sm][sn][rr];
    }
}

// ---------------------------------------------------------------- phase 2
// 32 persistent WGs x 256 threads. WG g owns h columns [g*16, g*16+16).
// Wave w: gate q = w>>1, row half mi = w&1 -> one 16x16 a-tile per wave.
// Wh slice (512K x 16 cols) lives in 16 register B-fragments for all steps.
__global__ __launch_bounds__(256) void k_lstm_rec(
    const __bf16* __restrict__ WhT, const float* __restrict__ xW,
    __bf16* __restrict__ hb0, __bf16* __restrict__ hb1,
    float* __restrict__ out, unsigned* __restrict__ bar) {
  __shared__ __bf16 hl[32 * 520];       // staged h (32x512, stride 520 = TDM pad)
  __shared__ float a_buf[4 * 32 * 16];  // gate pre-activations
  __shared__ float c_buf[32 * 16];      // persistent cell state (this WG's cols)

  const int tid  = threadIdx.x;
  const int lane = tid & 31, w = tid >> 5;
  const int ln = lane & 15, hi = lane >> 4;
  const int q  = w >> 1;       // gate 0..3 (i,f,o,g)
  const int mi = w & 1;        // batch-row half
  const int g  = blockIdx.x;   // 0..31 column group

  const int colg = q * 512 + g * 16 + ln;  // global gate column

  v16bf bfrag[16];
#pragma unroll
  for (int kc = 0; kc < 16; ++kc) {
    BF32B u;
    const __bf16* p = WhT + (size_t)colg * 512 + kc * 32 + hi * 8;
    u.q[0] = *(const uint4*)p;
    u.q[1] = *(const uint4*)(p + 16);
    bfrag[kc] = u.v;
  }

  c_buf[tid] = 0.0f;
  c_buf[tid + 256] = 0.0f;

  for (int t = 0; t < 1024; ++t) {
    const __bf16* hsrc = (t & 1) ? hb1 : hb0;
    __bf16*       hdst = (t & 1) ? hb0 : hb1;
#if USE_TDM
    if (w == 0)  // DMA the 32x512 bf16 h matrix into LDS (padded rows)
      tdm_load_2d((uint64_t)(uintptr_t)hsrc, lds_off(hl), 512, 32, 512, TDM_KF2);
#else
    {  // stage h (32 KB) into LDS: 256 threads x 128 B
      int row = tid >> 3, seg = tid & 7;
      const uint4* s = (const uint4*)(hsrc + row * 512 + seg * 64);
      uint4* d = (uint4*)(hl + row * 520 + seg * 64);
#pragma unroll
      for (int j = 0; j < 8; ++j) d[j] = s[j];
    }
#endif
    // accumulator initialized with xW[:, t, colg] (C layout: rows hi*8+r, col ln)
    v8f acc;
#pragma unroll
    for (int rr = 0; rr < 8; ++rr) {
      int rowb = mi * 16 + hi * 8 + rr;
      acc[rr] = xW[((size_t)rowb * 1024 + t) * 2048 + colg];
    }
#if USE_TDM
    if (w == 0) __builtin_amdgcn_s_wait_tensorcnt(0);
#endif
    __syncthreads();
#pragma unroll
    for (int kc = 0; kc < 16; ++kc) {
      BF32B a;
      const __bf16* p = hl + (mi * 16 + ln) * 520 + kc * 32 + hi * 8;
      a.q[0] = *(const uint4*)p;
      a.q[1] = *(const uint4*)(p + 16);
      acc = __builtin_amdgcn_wmma_f32_16x16x32_bf16(
          false, a.v, false, bfrag[kc], (short)0, acc, false, false);
    }
#pragma unroll
    for (int rr = 0; rr < 8; ++rr)
      a_buf[q * 512 + (mi * 16 + hi * 8 + rr) * 16 + ln] = acc[rr];
    __syncthreads();
    // fused LSTM cell for this WG's 32x16 h-slice
#pragma unroll
    for (int e0 = 0; e0 < 2; ++e0) {
      int e = tid + e0 * 256;
      int row = e >> 4, col = e & 15;
      float ii = sigmf(a_buf[e]);
      float ff = sigmf(a_buf[512 + e]);
      float oo = sigmf(a_buf[1024 + e]);
      float gg = tanhf(a_buf[1536 + e]);
      float cn = ff * c_buf[e] + ii * gg;
      float hn = oo * tanhf(cn);
      c_buf[e] = cn;
      hdst[row * 512 + g * 16 + col] = (__bf16)hn;
      out[((size_t)row * 1024 + t) * 512 + g * 16 + col] = hn;
    }
    // grid-wide barrier (monotonic counter, no reset races)
    __threadfence();
    __syncthreads();
    if (tid == 0) {
      __hip_atomic_fetch_add(bar, 1u, __ATOMIC_ACQ_REL, __HIP_MEMORY_SCOPE_AGENT);
      unsigned target = (unsigned)(t + 1) * 32u;
      while (__hip_atomic_load(bar, __ATOMIC_ACQUIRE, __HIP_MEMORY_SCOPE_AGENT) < target)
        __builtin_amdgcn_s_sleep(1);
    }
    __syncthreads();
#if defined(__has_builtin)
#if __has_builtin(__builtin_amdgcn_s_cluster_barrier)
    // architecturally NOP when ClusterID==0 (non-cluster dispatch)
    __builtin_amdgcn_s_cluster_barrier();
#endif
#endif
  }
}

// ---------------------------------------------------------------- launch
extern "C" void kernel_launch(void* const* d_in, const int* in_sizes, int n_in,
                              void* d_out, int out_size, void* d_ws, size_t ws_size,
                              hipStream_t stream) {
  const float* x  = (const float*)d_in[0];  // (32,1024,512)
  const float* h0 = (const float*)d_in[1];  // (32,512)
  const float* Wx = (const float*)d_in[2];  // (512,2048)
  const float* Wh = (const float*)d_in[3];  // (512,2048)
  const float* b  = (const float*)d_in[4];  // (2048,)
  float* out = (float*)d_out;               // (32,1024,512) f32

  char* ws = (char*)d_ws;
  size_t off = 0;
  __bf16* xb  = (__bf16*)(ws + off); off += (size_t)32768 * 512 * 2;  // 32 MB
  __bf16* WxT = (__bf16*)(ws + off); off += (size_t)2048 * 512 * 2;   // 2 MB
  __bf16* WhT = (__bf16*)(ws + off); off += (size_t)2048 * 512 * 2;   // 2 MB
  __bf16* hb0 = (__bf16*)(ws + off); off += (size_t)32 * 512 * 2;
  __bf16* hb1 = (__bf16*)(ws + off); off += (size_t)32 * 512 * 2;
  unsigned* bar = (unsigned*)(ws + off); off += 256;
  float* xW = (float*)(ws + off); off += (size_t)32768 * 2048 * 4;    // 256 MB

  hipMemsetAsync(bar, 0, sizeof(unsigned), stream);
  k_cvt_bf16<<<(16777216 + 255) / 256, 256, 0, stream>>>(x, xb, 16777216);
  k_transpose512x2048<<<4096, 256, 0, stream>>>(Wx, WxT);
  k_transpose512x2048<<<4096, 256, 0, stream>>>(Wh, WhT);
  k_cvt_bf16<<<64, 256, 0, stream>>>(h0, hb0, 16384);
  dim3 g1(16, 256);
  k_gemm_xw<<<g1, 256, 0, stream>>>(xb, WxT, b, xW);
  k_lstm_rec<<<32, 256, 0, stream>>>(WhT, xW, hb0, hb1, out, bar);
}
